// EncoderLayer_59176059404392
// MI455X (gfx1250) — compile-verified, matrix operand order checked by reading, not run on hardware
//
#include <hip/hip_runtime.h>
#include <math.h>

#define BATCH 8192
#define CCH 12
#define DD 64

typedef __attribute__((ext_vector_type(16))) _Float16 v16h;
typedef __attribute__((ext_vector_type(8)))  float    v8f;

// ---------- math helpers ----------
__device__ __forceinline__ float frcp(float x)      { return __builtin_amdgcn_rcpf(x); }
__device__ __forceinline__ float siluf(float v)     { return v * frcp(1.0f + __expf(-v)); }
__device__ __forceinline__ float softplusf(float v) { return v > 20.0f ? v : log1pf(__expf(v)); }
__device__ __forceinline__ float geluf(float v)     { return 0.5f * v * (1.0f + erff(v * 0.70710678118f)); }

// ---------- WMMA helpers ----------
// 16-bit A-matrix fragment (ISA 7.12.2): lane l holds row M = l%16; K-halves:
//  f[j] = row[k0 + 8*(l/16) + j] (j<8), f[j] = row[k0 + 16 + 8*(l/16) + (j-8)] (j>=8)
// B (K x N) from row-major W[N][K]: B[k][n] = W[n][k] -> same loader, row0 = n-tile base.
__device__ __forceinline__ v16h ldfrag(const _Float16* __restrict__ base, int row0, int stride, int k0) {
  const int lane = threadIdx.x & 31;
  const int m  = lane & 15;
  const int hi = lane >> 4;
  const _Float16* p = base + (size_t)(row0 + m) * stride + k0 + hi * 8;
  v16h f;
#pragma unroll
  for (int j = 0; j < 8; ++j) f[j] = p[j];
#pragma unroll
  for (int j = 0; j < 8; ++j) f[8 + j] = p[16 + j];
  return f;
}

__device__ __forceinline__ v8f wmma16(v16h a, v16h b, v8f c) {
  return __builtin_amdgcn_wmma_f32_16x16x32_f16(false, a, false, b, (short)0, c, false, false);
}

// vectorized f32 row -> f16 LDS row copy helper (16B global loads)
__device__ __forceinline__ void cp_row4(_Float16* dst, const float* src, int q) {
  const float4 v = ((const float4*)src)[q];
  dst[4 * q + 0] = (_Float16)v.x;
  dst[4 * q + 1] = (_Float16)v.y;
  dst[4 * q + 2] = (_Float16)v.z;
  dst[4 * q + 3] = (_Float16)v.w;
}

// =====================================================================
// Kernel 1: per-channel Mamba block -> all_out (B,C,D) f32
// =====================================================================
#define BT1 128
#define ST1 72

__global__ __launch_bounds__(256) void k_mamba(
    const float* __restrict__ x, const float* __restrict__ in_proj_w,
    const float* __restrict__ conv_w, const float* __restrict__ conv_b,
    const float* __restrict__ x_proj_w, const float* __restrict__ dt_proj_w,
    const float* __restrict__ dt_proj_b, const float* __restrict__ D_ssm,
    const float* __restrict__ out_proj_w, const float* __restrict__ channel_weights,
    float* __restrict__ allout)
{
  extern __shared__ char smem[];
  _Float16* sX   = (_Float16*)smem;        // 128x72 : x (f16), later reused as y
  _Float16* sWin = sX   + BT1 * ST1;       // 128x72
  _Float16* sXc  = sWin + 128 * ST1;       // 128x72
  _Float16* sZ   = sXc  + BT1 * ST1;       // 128x72
  _Float16* sXp  = sZ   + BT1 * ST1;       // 16x72
  _Float16* sWo  = sXp  + 16  * ST1;       // 64x72
  float* sProj = (float*)(sWo + 64 * ST1); // 128x17
  float* sScal = sProj + 128 * 17;         // 128
  float* sCw   = sScal + 128;              // 64
  float* sCb   = sCw + 64;                 // 64
  float* sDssm = sCb + 64;                 // 64
  float* sDtb  = sDssm + 64;               // 64
  float* sDtw  = sDtb + 64;                // 64x4

  const int c   = blockIdx.y;
  const int r0  = blockIdx.x * BT1;
  const int tid = threadIdx.x;
  const int w    = tid >> 5;
  const int lane = tid & 31;
  const int idx  = lane & 15;
  const int hi   = lane >> 4;

  // ---- stage activations + weights into LDS as f16 (float4 global reads) ----
  for (int i = tid; i < BT1 * 16; i += 256) {   // x rows
    int r = i >> 4, q = i & 15;
    cp_row4(sX + r * ST1, x + ((size_t)(r0 + r) * CCH + c) * DD, q);
  }
  for (int i = tid; i < 128 * 16; i += 256) {   // in_proj rows
    int r = i >> 4, q = i & 15;
    cp_row4(sWin + r * ST1, in_proj_w + ((size_t)c * 128 + r) * 64, q);
  }
  for (int i = tid; i < 16 * 16; i += 256) {    // x_proj rows
    int r = i >> 4, q = i & 15;
    cp_row4(sXp + r * ST1, x_proj_w + ((size_t)c * 16 + r) * 64, q);
  }
  for (int i = tid; i < 64 * 16; i += 256) {    // out_proj rows
    int r = i >> 4, q = i & 15;
    cp_row4(sWo + r * ST1, out_proj_w + ((size_t)c * 64 + r) * 64, q);
  }
  if (tid < 64) {
    sCw[tid]   = conv_w[((size_t)c * 64 + tid) * 2 + 1];
    sCb[tid]   = conv_b[c * 64 + tid];
    sDssm[tid] = D_ssm[c * 64 + tid];
    sDtb[tid]  = dt_proj_b[c * 64 + tid];
  }
  for (int i = tid; i < 256; i += 256) sDtw[i] = dt_proj_w[(size_t)c * 256 + i];
  const float chw = channel_weights[c];
  __syncthreads();

  // ---- GEMM1: XZ = X(128x64) @ Win^T(64x128); A-fragments hoisted ----
  {
    const v16h a0 = ldfrag(sX, 16 * w, ST1, 0);
    const v16h a1 = ldfrag(sX, 16 * w, ST1, 32);
    for (int j = 0; j < 8; ++j) {
      v8f acc = {};
      acc = wmma16(a0, ldfrag(sWin, 16 * j, ST1, 0),  acc);
      acc = wmma16(a1, ldfrag(sWin, 16 * j, ST1, 32), acc);
      const int n = 16 * j + idx;
#pragma unroll
      for (int r = 0; r < 8; ++r) {
        const int m = 16 * w + 8 * hi + r;
        const float v = acc[r];
        if (n < 64) sXc[m * ST1 + n]        = (_Float16)siluf(v * sCw[n] + sCb[n]);
        else        sZ [m * ST1 + (n - 64)] = (_Float16)siluf(v);
      }
    }
  }
  __syncthreads();

  // ---- GEMM2: proj = xc(128x64) @ Xp^T(64x16) ----
  {
    v8f acc = {};
    acc = wmma16(ldfrag(sXc, 16 * w, ST1, 0),  ldfrag(sXp, 0, ST1, 0),  acc);
    acc = wmma16(ldfrag(sXc, 16 * w, ST1, 32), ldfrag(sXp, 0, ST1, 32), acc);
#pragma unroll
    for (int r = 0; r < 8; ++r) sProj[(16 * w + 8 * hi + r) * 17 + idx] = acc[r];
  }
  __syncthreads();

  // ---- scal[row] = dot(B_s, C_s) ----
  if (tid < BT1) {
    float s = 0.0f;
#pragma unroll
    for (int i = 0; i < 6; ++i) s += sProj[tid * 17 + 4 + i] * sProj[tid * 17 + 10 + i];
    sScal[tid] = s;
  }
  __syncthreads();

  // ---- y = ((softplus(dt)*xc)*scal + Dssm*xc) * silu(z); reuse sX ----
  for (int i = tid; i < BT1 * 64; i += 256) {
    const int r = i >> 6, d = i & 63;
    float dtv = sDtb[d];
#pragma unroll
    for (int q = 0; q < 4; ++q) dtv += sProj[r * 17 + q] * sDtw[d * 4 + q];
    dtv = softplusf(dtv);
    const float xcv = (float)sXc[r * ST1 + d];
    float y = (dtv * xcv) * sScal[r] + sDssm[d] * xcv;
    y *= (float)sZ[r * ST1 + d];
    sX[r * ST1 + d] = (_Float16)y;
  }
  __syncthreads();

  // ---- GEMM3: all_out = cw * (y @ Wo^T) + x ----
  {
    const v16h a0 = ldfrag(sX, 16 * w, ST1, 0);
    const v16h a1 = ldfrag(sX, 16 * w, ST1, 32);
    for (int j = 0; j < 4; ++j) {
      v8f acc = {};
      acc = wmma16(a0, ldfrag(sWo, 16 * j, ST1, 0),  acc);
      acc = wmma16(a1, ldfrag(sWo, 16 * j, ST1, 32), acc);
      const int n = 16 * j + idx;
#pragma unroll
      for (int r = 0; r < 8; ++r) {
        const int m = 16 * w + 8 * hi + r;
        const size_t gi = ((size_t)(r0 + m) * CCH + c) * DD + n;
        allout[gi] = chw * acc[r] + x[gi];
      }
    }
  }
}

// =====================================================================
// Kernel 2: encoder over E=C=12, S=D=64 (heads=4, hd=3). 1 wave = 1 batch.
// =====================================================================
__global__ __launch_bounds__(256) void k_enc1(
    const float* __restrict__ allout,
    const float* __restrict__ a1w, const float* __restrict__ a1b,
    const float* __restrict__ a1ow, const float* __restrict__ a1ob,
    const float* __restrict__ n1g, const float* __restrict__ n1b,
    const float* __restrict__ m1w1, const float* __restrict__ m1b1,
    const float* __restrict__ m1w2, const float* __restrict__ m1b2,
    float* __restrict__ x2buf)
{
  extern __shared__ char smem[];
  float* Tall = (float*)smem;           // 8 x 64x13
  float* Qall = Tall + 8 * 832;         // 8 x 64x37
  float* Oall = Qall + 8 * 2368;        // 8 x 64x13
  float* sWin = Oall + 8 * 832;         // 36x12
  float* sBin = sWin + 432;
  float* sOw  = sBin + 36;              // 12x12
  float* sOb  = sOw + 144;
  float* sG   = sOb + 12;
  float* sBn  = sG + 12;
  float* sW1  = sBn + 12;               // 24x12
  float* sB1  = sW1 + 288;
  float* sW2  = sB1 + 24;               // 12x24
  float* sB2  = sW2 + 288;

  const int tid = threadIdx.x;
  const int w = tid >> 5, lane = tid & 31;
  float* T = Tall + w * 832;   // [64][13]
  float* Q = Qall + w * 2368;  // [64][37]
  float* O = Oall + w * 832;   // [64][13]

  for (int i = tid; i < 432; i += 256) sWin[i] = a1w[i];
  if (tid < 36) sBin[tid] = a1b[tid];
  for (int i = tid; i < 144; i += 256) sOw[i] = a1ow[i];
  if (tid < 12) { sOb[tid] = a1ob[tid]; sG[tid] = n1g[tid]; sBn[tid] = n1b[tid]; sB2[tid] = m1b2[tid]; }
  for (int i = tid; i < 288; i += 256) { sW1[i] = m1w1[i]; sW2[i] = m1w2[i]; }
  if (tid < 24) sB1[tid] = m1b1[tid];

  const int b = blockIdx.x * 8 + w;
  // t[s][e] = allout[b][e][s]  (e-major loop -> lanes read consecutive s: coalesced)
  for (int i = lane; i < 64 * 12; i += 32) {
    int e = i >> 6, s = i & 63;
    T[s * 13 + e] = allout[((size_t)b * CCH + e) * DD + s];
  }
  __syncthreads();

  // qkv = t @ Win^T + b : 64x36
  for (int i = lane; i < 64 * 36; i += 32) {
    int s = i / 36, o = i % 36;
    float acc = sBin[o];
#pragma unroll
    for (int e = 0; e < 12; ++e) acc += T[s * 13 + e] * sWin[o * 12 + e];
    Q[s * 37 + o] = acc;
  }
  __syncthreads();

  // attention: 4 heads x 64 queries = 256 tasks / wave
  const float scale = 0.57735026919f; // 1/sqrt(3)
  for (int p = lane; p < 256; p += 32) {
    const int h = p >> 6, s = p & 63;
    const float q0 = Q[s * 37 + h * 3 + 0] * scale;
    const float q1 = Q[s * 37 + h * 3 + 1] * scale;
    const float q2 = Q[s * 37 + h * 3 + 2] * scale;
    float sc[64];
    float mx = -1e30f;
#pragma unroll
    for (int t = 0; t < 64; ++t) {
      const float* kr = &Q[t * 37 + 12 + h * 3];
      float v = q0 * kr[0] + q1 * kr[1] + q2 * kr[2];
      sc[t] = v; mx = fmaxf(mx, v);
    }
    float sum = 0.0f;
#pragma unroll
    for (int t = 0; t < 64; ++t) { sc[t] = __expf(sc[t] - mx); sum += sc[t]; }
    const float inv = frcp(sum);
    float o0 = 0, o1 = 0, o2 = 0;
#pragma unroll
    for (int t = 0; t < 64; ++t) {
      const float* vr = &Q[t * 37 + 24 + h * 3];
      o0 += sc[t] * vr[0]; o1 += sc[t] * vr[1]; o2 += sc[t] * vr[2];
    }
    O[s * 13 + h * 3 + 0] = o0 * inv;
    O[s * 13 + h * 3 + 1] = o1 * inv;
    O[s * 13 + h * 3 + 2] = o2 * inv;
  }
  __syncthreads();

  // out proj + residual -> P (reuse Q region, stride 13)
  float* P = Q;
  for (int i = lane; i < 64 * 12; i += 32) {
    int s = i / 12, e = i % 12;
    float acc = sOb[e];
#pragma unroll
    for (int e2 = 0; e2 < 12; ++e2) acc += O[s * 13 + e2] * sOw[e * 12 + e2];
    P[s * 13 + e] = T[s * 13 + e] + acc;
  }
  __syncthreads();

  // LN1 (over 12) -> A1 into T
  for (int s = lane; s < 64; s += 32) {
    float mu = 0;
#pragma unroll
    for (int e = 0; e < 12; ++e) mu += P[s * 13 + e];
    mu *= (1.0f / 12.0f);
    float var = 0;
#pragma unroll
    for (int e = 0; e < 12; ++e) { float d = P[s * 13 + e] - mu; var += d * d; }
    var *= (1.0f / 12.0f);
    const float rs = rsqrtf(var + 1e-5f);
#pragma unroll
    for (int e = 0; e < 12; ++e)
      T[s * 13 + e] = (P[s * 13 + e] - mu) * rs * sG[e] + sBn[e];
  }
  __syncthreads();

  // MLP 12->24->12 (gelu exact), residual -> A2 into O
  for (int s = lane; s < 64; s += 32) {
    float hbuf[24];
#pragma unroll
    for (int j = 0; j < 24; ++j) {
      float hv = sB1[j];
#pragma unroll
      for (int e = 0; e < 12; ++e) hv += T[s * 13 + e] * sW1[j * 12 + e];
      hbuf[j] = geluf(hv);
    }
#pragma unroll
    for (int e = 0; e < 12; ++e) {
      float acc = sB2[e];
#pragma unroll
      for (int j = 0; j < 24; ++j) acc += hbuf[j] * sW2[e * 24 + j];
      O[s * 13 + e] = T[s * 13 + e] + acc;
    }
  }
  __syncthreads();

  // LN (norm1 again) -> x2[b][e][s]
  for (int s = lane; s < 64; s += 32) {
    float mu = 0;
#pragma unroll
    for (int e = 0; e < 12; ++e) mu += O[s * 13 + e];
    mu *= (1.0f / 12.0f);
    float var = 0;
#pragma unroll
    for (int e = 0; e < 12; ++e) { float d = O[s * 13 + e] - mu; var += d * d; }
    var *= (1.0f / 12.0f);
    const float rs = rsqrtf(var + 1e-5f);
#pragma unroll
    for (int e = 0; e < 12; ++e)
      x2buf[((size_t)b * CCH + e) * DD + s] = (O[s * 13 + e] - mu) * rs * sG[e] + sBn[e];
  }
}

// =====================================================================
// Kernel 3: encoder over E=D=64, S=C=12 (heads=8, hd=8).
// =====================================================================
#define GB3 8
#define RW3 96
#define ST3 72
#define QS3 200

__global__ __launch_bounds__(256) void k_enc2(
    const float* __restrict__ x2buf,
    const float* __restrict__ a2w, const float* __restrict__ a2b,
    const float* __restrict__ a2ow, const float* __restrict__ a2ob,
    const float* __restrict__ n2g, const float* __restrict__ n2b,
    const float* __restrict__ m2w1, const float* __restrict__ m2b1,
    const float* __restrict__ m2w2, const float* __restrict__ m2b2,
    float* __restrict__ out)
{
  extern __shared__ char smem[];
  _Float16* sX   = (_Float16*)smem;     // 96x72
  _Float16* sY   = sX   + RW3 * ST3;    // 96x72
  _Float16* sQKV = sY   + RW3 * ST3;    // 96x200
  _Float16* sWq  = sQKV + RW3 * QS3;    // 192x72
  _Float16* sWo  = sWq  + 192 * ST3;    // 64x72
  _Float16* sW1  = sWo  + 64  * ST3;    // 128x72
  _Float16* sW2  = sW1  + 128 * ST3;    // 64x136
  float* fInB = (float*)(sW2 + 64 * 136); // 192
  float* fOB  = fInB + 192;
  float* fG   = fOB + 64;
  float* fBt  = fG + 64;
  float* fM1b = fBt + 64;                 // 128
  float* fM2b = fM1b + 128;

  const int tid = threadIdx.x;
  const int w = tid >> 5, lane = tid & 31, idx = lane & 15, hi = lane >> 4;
  const int b0 = blockIdx.x * GB3;

  for (int i = tid; i < RW3 * 16; i += 256) {
    int r = i >> 4, q = i & 15;
    cp_row4(sX + r * ST3, x2buf + ((size_t)(b0 + r / 12) * CCH + (r % 12)) * DD, q);
  }
  for (int i = tid; i < 192 * 16; i += 256) {
    int r = i >> 4, q = i & 15;
    cp_row4(sWq + r * ST3, a2w + (size_t)r * 64, q);
  }
  for (int i = tid; i < 64 * 16; i += 256) {
    int r = i >> 4, q = i & 15;
    cp_row4(sWo + r * ST3, a2ow + (size_t)r * 64, q);
  }
  for (int i = tid; i < 128 * 16; i += 256) {
    int r = i >> 4, q = i & 15;
    cp_row4(sW1 + r * ST3, m2w1 + (size_t)r * 64, q);
  }
  for (int i = tid; i < 64 * 32; i += 256) {
    int r = i >> 5, q = i & 31;
    cp_row4(sW2 + r * 136, m2w2 + (size_t)r * 128, q);
  }
  if (tid < 192) fInB[tid] = a2b[tid];
  if (tid < 64) { fOB[tid] = a2ob[tid]; fG[tid] = n2g[tid]; fBt[tid] = n2b[tid]; fM2b[tid] = m2b2[tid]; }
  if (tid < 128) fM1b[tid] = m2b1[tid];
  __syncthreads();

  // qkv: (96x64)@(64x192) -> 6x12 = 72 tiles
  for (int t = w; t < 72; t += 8) {
    const int i = t / 12, j = t % 12;
    v8f acc = {};
    acc = wmma16(ldfrag(sX, 16 * i, ST3, 0),  ldfrag(sWq, 16 * j, ST3, 0),  acc);
    acc = wmma16(ldfrag(sX, 16 * i, ST3, 32), ldfrag(sWq, 16 * j, ST3, 32), acc);
    const int n = 16 * j + idx;
#pragma unroll
    for (int r = 0; r < 8; ++r) {
      const int m = 16 * i + 8 * hi + r;
      sQKV[m * QS3 + n] = (_Float16)(acc[r] + fInB[n]);
    }
  }
  __syncthreads();

  // attention core: (bi<8, h<8, s<12) = 768 tasks
  const float scale = 0.35355339059f; // 1/sqrt(8)
  for (int p = tid; p < 768; p += 256) {
    const int bi = p / 96, rem = p % 96, h = rem / 12, s = rem % 12;
    const int qrow = bi * 12 + s;
    float q[8];
#pragma unroll
    for (int d = 0; d < 8; ++d) q[d] = (float)sQKV[qrow * QS3 + h * 8 + d] * scale;
    float sc[12];
    float mx = -1e30f;
#pragma unroll
    for (int t = 0; t < 12; ++t) {
      const _Float16* kr = &sQKV[(bi * 12 + t) * QS3 + 64 + h * 8];
      float v = 0;
#pragma unroll
      for (int d = 0; d < 8; ++d) v += q[d] * (float)kr[d];
      sc[t] = v; mx = fmaxf(mx, v);
    }
    float sum = 0;
#pragma unroll
    for (int t = 0; t < 12; ++t) { sc[t] = __expf(sc[t] - mx); sum += sc[t]; }
    const float inv = frcp(sum);
#pragma unroll
    for (int d = 0; d < 8; ++d) {
      float o = 0;
#pragma unroll
      for (int t = 0; t < 12; ++t) o += sc[t] * (float)sQKV[(bi * 12 + t) * QS3 + 128 + h * 8 + d];
      sY[qrow * ST3 + h * 8 + d] = (_Float16)(o * inv);
    }
  }
  __syncthreads();

  // out-proj + residual -> pre-LN into sQKV[:, :64]
  for (int t = w; t < 24; t += 8) {
    const int i = t / 4, j = t % 4;
    v8f acc = {};
    acc = wmma16(ldfrag(sY, 16 * i, ST3, 0),  ldfrag(sWo, 16 * j, ST3, 0),  acc);
    acc = wmma16(ldfrag(sY, 16 * i, ST3, 32), ldfrag(sWo, 16 * j, ST3, 32), acc);
    const int n = 16 * j + idx;
#pragma unroll
    for (int r = 0; r < 8; ++r) {
      const int m = 16 * i + 8 * hi + r;
      sQKV[m * QS3 + n] = (_Float16)(acc[r] + fOB[n] + (float)sX[m * ST3 + n]);
    }
  }
  __syncthreads();

  // LN(64) -> new activation in sX
  if (tid < RW3) {
    const int row = tid;
    float mu = 0;
    for (int d = 0; d < 64; ++d) mu += (float)sQKV[row * QS3 + d];
    mu *= (1.0f / 64.0f);
    float var = 0;
    for (int d = 0; d < 64; ++d) { float dd = (float)sQKV[row * QS3 + d] - mu; var += dd * dd; }
    var *= (1.0f / 64.0f);
    const float rs = rsqrtf(var + 1e-5f);
    for (int d = 0; d < 64; ++d)
      sX[row * ST3 + d] = (_Float16)(((float)sQKV[row * QS3 + d] - mu) * rs * fG[d] + fBt[d]);
  }
  __syncthreads();

  // MLP in: (96x64)@(64x128), gelu -> hidden in sQKV
  for (int t = w; t < 48; t += 8) {
    const int i = t / 8, j = t % 8;
    v8f acc = {};
    acc = wmma16(ldfrag(sX, 16 * i, ST3, 0),  ldfrag(sW1, 16 * j, ST3, 0),  acc);
    acc = wmma16(ldfrag(sX, 16 * i, ST3, 32), ldfrag(sW1, 16 * j, ST3, 32), acc);
    const int n = 16 * j + idx;
#pragma unroll
    for (int r = 0; r < 8; ++r) {
      const int m = 16 * i + 8 * hi + r;
      sQKV[m * QS3 + n] = (_Float16)geluf(acc[r] + fM1b[n]);
    }
  }
  __syncthreads();

  // MLP out: (96x128)@(128x64) + residual -> sY, K=128
  for (int t = w; t < 24; t += 8) {
    const int i = t / 4, j = t % 4;
    v8f acc = {};
#pragma unroll
    for (int kk = 0; kk < 4; ++kk) {
      v16h a = ldfrag(sQKV, 16 * i, QS3, 32 * kk);
      v16h b = ldfrag(sW2,  16 * j, 136, 32 * kk);
      acc = wmma16(a, b, acc);
    }
    const int n = 16 * j + idx;
#pragma unroll
    for (int r = 0; r < 8; ++r) {
      const int m = 16 * i + 8 * hi + r;
      sY[m * ST3 + n] = (_Float16)(acc[r] + fM2b[n] + (float)sX[m * ST3 + n]);
    }
  }
  __syncthreads();

  // final LN(64) -> out[b][c][d]
  if (tid < RW3) {
    const int row = tid;
    const int bi = row / 12, cc = row % 12;
    float mu = 0;
    for (int d = 0; d < 64; ++d) mu += (float)sY[row * ST3 + d];
    mu *= (1.0f / 64.0f);
    float var = 0;
    for (int d = 0; d < 64; ++d) { float dd = (float)sY[row * ST3 + d] - mu; var += dd * dd; }
    var *= (1.0f / 64.0f);
    const float rs = rsqrtf(var + 1e-5f);
    float* op = out + ((size_t)(b0 + bi) * CCH + cc) * DD;
    for (int d = 0; d < 64; ++d)
      op[d] = ((float)sY[row * ST3 + d] - mu) * rs * fG[d] + fBt[d];
  }
}

// =====================================================================
extern "C" void kernel_launch(void* const* d_in, const int* in_sizes, int n_in,
                              void* d_out, int out_size, void* d_ws, size_t ws_size,
                              hipStream_t stream) {
  (void)in_sizes; (void)n_in; (void)out_size; (void)ws_size;
  const float* x          = (const float*)d_in[0];
  const float* in_proj_w  = (const float*)d_in[2];
  const float* conv_w     = (const float*)d_in[3];
  const float* conv_b     = (const float*)d_in[4];
  const float* x_proj_w   = (const float*)d_in[5];
  const float* dt_proj_w  = (const float*)d_in[6];
  const float* dt_proj_b  = (const float*)d_in[7];
  const float* D_ssm      = (const float*)d_in[9];
  const float* out_proj_w = (const float*)d_in[10];
  const float* ch_w       = (const float*)d_in[11];
  const float* a1w  = (const float*)d_in[12];
  const float* a1b  = (const float*)d_in[13];
  const float* a1ow = (const float*)d_in[14];
  const float* a1ob = (const float*)d_in[15];
  const float* a2w  = (const float*)d_in[16];
  const float* a2b  = (const float*)d_in[17];
  const float* a2ow = (const float*)d_in[18];
  const float* a2ob = (const float*)d_in[19];
  const float* n1g  = (const float*)d_in[20];
  const float* n1b  = (const float*)d_in[21];
  const float* n2g  = (const float*)d_in[22];
  const float* n2b  = (const float*)d_in[23];
  const float* m1w1 = (const float*)d_in[24];
  const float* m1b1 = (const float*)d_in[25];
  const float* m1w2 = (const float*)d_in[26];
  const float* m1b2 = (const float*)d_in[27];
  const float* m2w1 = (const float*)d_in[28];
  const float* m2b1 = (const float*)d_in[29];
  const float* m2w2 = (const float*)d_in[30];
  const float* m2b2 = (const float*)d_in[31];

  float* allout = (float*)d_ws;
  float* x2buf  = allout + (size_t)BATCH * CCH * DD;

  constexpr size_t SM1 = 2 * (size_t)(4 * 128 * ST1 + 16 * ST1 + 64 * ST1)
                       + 4 * (size_t)(128 * 17 + 128 + 64 * 4 + 256);
  constexpr size_t SM2 = 4 * (size_t)(8 * (832 + 2368 + 832)
                       + 432 + 36 + 144 + 12 + 12 + 12 + 288 + 24 + 288 + 12);
  constexpr size_t SM3 = 2 * (size_t)(2 * RW3 * ST3 + RW3 * QS3 + 192 * ST3
                       + 64 * ST3 + 128 * ST3 + 64 * 136)
                       + 4 * (size_t)(192 + 64 + 64 + 64 + 128 + 64);

  dim3 g1(BATCH / BT1, CCH);
  k_mamba<<<g1, 256, SM1, stream>>>(x, in_proj_w, conv_w, conv_b, x_proj_w,
                                    dt_proj_w, dt_proj_b, D_ssm, out_proj_w, ch_w, allout);
  k_enc1<<<BATCH / 8, 256, SM2, stream>>>(allout, a1w, a1b, a1ow, a1ob,
                                          n1g, n1b, m1w1, m1b1, m1w2, m1b2, x2buf);
  k_enc2<<<BATCH / GB3, 256, SM3, stream>>>(x2buf, a2w, a2b, a2ow, a2ob,
                                            n2g, n2b, m2w1, m2b1, m2w2, m2b2, (float*)d_out);
}
